// VQVAE_62551903699304
// MI455X (gfx1250) — matrix operand order repaired and checked by reading, not compiled, not run on hardware
//
#include <hip/hip_runtime.h>

// ---------------------------------------------------------------------------
// VQ-VAE forward for MI455X (gfx1250, wave32).
//  - bf16 NHWC activations: A fragments = two 16B vector loads per lane.
//  - Weights repacked once into WMMA B-fragment layout; per kernel tap the
//    workgroup stages them into LDS (ds_load in the hot loop).
//  - 4 waves / block share B; each wave: 32-pixel x 64-channel tile ->
//    8 x v_wmma_f32_16x16x32_bf16 per K-chunk (f32 accumulate).
// ---------------------------------------------------------------------------

typedef __attribute__((ext_vector_type(16))) __bf16 v16bf;
typedef __attribute__((ext_vector_type(8)))  __bf16 v8bf;
typedef __attribute__((ext_vector_type(8)))  float  v8f;

#define EMB_D 64
#define EMB_K 128

// ---------------------------------------------------------------------------
// Weight repack: f32 (OIHW conv, or (Cin,Cout,5,5) convT) -> bf16 fragments.
// Layout: [kk][cb][nb][lane][16], value = w[co=nb*16+l15][ci=cb*32+e+16*hf].
// ---------------------------------------------------------------------------
__global__ void repack_w(const float* __restrict__ w, __bf16* __restrict__ o,
                         int total, int Cin, int Cout, int transposed)
{
    int idx = blockIdx.x * blockDim.x + threadIdx.x;
    if (idx >= total) return;
    const int e    = idx & 15;
    const int lane = (idx >> 4) & 31;
    const int tile = idx >> 9;
    const int nCo16 = Cout >> 4;
    const int nCi32 = Cin >> 5;
    const int nb = tile % nCo16;
    const int cb = (tile / nCo16) % nCi32;
    const int kk = tile / (nCo16 * nCi32);
    const int hf  = lane >> 4;
    const int l15 = lane & 15;
    const int ci = cb * 32 + e + 16 * hf;
    const int co = nb * 16 + l15;
    const float v = transposed ? w[((size_t)ci * Cout + co) * 25 + kk]
                               : w[((size_t)co * Cin + ci) * 25 + kk];
    o[idx] = (__bf16)v;
}

// ---------------------------------------------------------------------------
// WMMA implicit-GEMM conv / conv-transpose.
// Block = 4 waves; all share one 64-channel N-tile (LDS-staged B fragments),
// each wave owns a 32-pixel M-tile (2 x M16), K = 32 channels per step.
// ---------------------------------------------------------------------------
template <int TRANSPOSED, int OUT_F32, int STRIDE>
__global__ __launch_bounds__(128)
void conv_wmma(const __bf16* __restrict__ in, const __bf16* __restrict__ wrep,
               const float* __restrict__ bias, void* __restrict__ outv,
               int Cin, int Cout, int Hin, int Win, int Hout, int Wout, int pad)
{
    __shared__ __align__(32) __bf16 smem[8192];   // up to (Cin=128): 4 chunks x 4KB

    const int tid  = threadIdx.x;   // 0..127
    const int wv   = tid >> 5;      // wave in block: 0..3
    const int lane = tid & 31;
    const int hf   = lane >> 4;
    const int l15  = lane & 15;

    const int n      = blockIdx.z;
    const int coBase = blockIdx.y * 64;
    const int co16   = coBase >> 4;
    const int pBase  = (blockIdx.x * 4 + wv) * 32;

    const int HWout = Hout * Wout;
    const int HWin  = Hin * Win;
    const int nCi32 = Cin >> 5;
    const int nCo16 = Cout >> 4;

    // Two pixel groups (M-tiles); per-lane M = l15 within each tile.
    const int  pix0 = pBase + l15,        pix1 = pBase + 16 + l15;
    const bool pv0  = (pix0 < HWout),     pv1  = (pix1 < HWout);
    const int  oy0  = pv0 ? (pix0 / Wout) : 0;
    const int  ox0  = pv0 ? (pix0 - oy0 * Wout) : 0;
    const int  oy1  = pv1 ? (pix1 / Wout) : 0;
    const int  ox1  = pv1 ? (pix1 - oy1 * Wout) : 0;

    const __bf16* inN = in + (size_t)n * HWin * Cin;

    auto srcIdx = [&](int o, int k, int lim, int& i) -> bool {
        if (!TRANSPOSED) {
            i = o * STRIDE - pad + k;
            return (i >= 0) && (i < lim);
        } else if (STRIDE == 1) {
            i = o + pad - k;
            return (i >= 0) && (i < lim);
        } else {
            const int t = o + pad - k;
            i = t >> 1;
            return (t >= 0) && !(t & 1) && (i < lim);
        }
    };

    v8f acc0[4] = {};
    v8f acc1[4] = {};

    for (int ky = 0; ky < 5; ++ky) {
        int iy0, iy1;
        const bool vy0 = srcIdx(oy0, ky, Hin, iy0);
        const bool vy1 = srcIdx(oy1, ky, Hin, iy1);
        for (int kx = 0; kx < 5; ++kx) {
            int ix0, ix1;
            const bool a0Ok = pv0 && vy0 && srcIdx(ox0, kx, Win, ix0);
            const bool a1Ok = pv1 && vy1 && srcIdx(ox1, kx, Win, ix1);
            const __bf16* aPix0 = inN + (size_t)(iy0 * Win + ix0) * Cin + hf * 16;
            const __bf16* aPix1 = inN + (size_t)(iy1 * Win + ix1) * Cin + hf * 16;
            const int kk = ky * 5 + kx;

            // ---- Stage this tap's B fragments into LDS (4KB per K-chunk):
            // each thread copies one 32B fragment slice per chunk.
            __syncthreads();   // previous iteration done reading smem
            const __bf16* wTap = wrep + (((size_t)kk * nCi32) * nCo16 + co16) * 512;
            for (int cb = 0; cb < nCi32; ++cb) {
                *(v16bf*)(smem + cb * 2048 + tid * 16) =
                    *(const v16bf*)(wTap + (size_t)cb * nCo16 * 512 + tid * 16);
            }
            // Warm L2 for the next tap while we compute this one.
            if (kk < 24)
                __builtin_prefetch(wTap + (size_t)nCi32 * nCo16 * 512 + tid * 16, 0, 1);
            __syncthreads();   // smem ready

            for (int cb = 0; cb < nCi32; ++cb) {
                // ---- Two A fragments (bf16 NHWC: 2 x 16B vector loads each)
                v16bf a0, a1;
                {
                    v8bf lo0 = {}, hi0 = {}, lo1 = {}, hi1 = {};
                    if (a0Ok) {
                        const __bf16* p = aPix0 + cb * 32;
                        lo0 = *(const v8bf*)(p);
                        hi0 = *(const v8bf*)(p + 8);
                    }
                    if (a1Ok) {
                        const __bf16* p = aPix1 + cb * 32;
                        lo1 = *(const v8bf*)(p);
                        hi1 = *(const v8bf*)(p + 8);
                    }
#pragma unroll
                    for (int e = 0; e < 8; ++e) {
                        a0[e] = lo0[e]; a0[e + 8] = hi0[e];
                        a1[e] = lo1[e]; a1[e + 8] = hi1[e];
                    }
                }
                // ---- 4 B fragments from LDS; each feeds 2 WMMAs
                const __bf16* bBase = smem + cb * 2048 + lane * 16;
#pragma unroll
                for (int nb = 0; nb < 4; ++nb) {
                    v16bf b = *(const v16bf*)(bBase + nb * 512);
                    acc0[nb] = __builtin_amdgcn_wmma_f32_16x16x32_bf16(
                        false, a0, false, b, (short)0, acc0[nb], false, false);
                    acc1[nb] = __builtin_amdgcn_wmma_f32_16x16x32_bf16(
                        false, a1, false, b, (short)0, acc1[nb], false, false);
                }
            }
        }
    }

    // ---- Store: D element r at lane l: M = r + 8*hf, N = l15
#pragma unroll
    for (int nb = 0; nb < 4; ++nb) {
        const int co = coBase + nb * 16 + l15;
        const float bv = bias[co];
        if (OUT_F32) {
            float* out = (float*)outv + ((size_t)n * Cout + co) * HWout;
#pragma unroll
            for (int r = 0; r < 8; ++r) {
                const int p0 = pBase + r + hf * 8;
                const int p1 = p0 + 16;
                if (p0 < HWout) out[p0] = fmaxf(acc0[nb][r] + bv, 0.0f);
                if (p1 < HWout) out[p1] = fmaxf(acc1[nb][r] + bv, 0.0f);
            }
        } else {
            __bf16* out = (__bf16*)outv + (size_t)n * HWout * Cout + co;
#pragma unroll
            for (int r = 0; r < 8; ++r) {
                const int p0 = pBase + r + hf * 8;
                const int p1 = p0 + 16;
                if (p0 < HWout) out[(size_t)p0 * Cout] = (__bf16)fmaxf(acc0[nb][r] + bv, 0.0f);
                if (p1 < HWout) out[(size_t)p1 * Cout] = (__bf16)fmaxf(acc1[nb][r] + bv, 0.0f);
            }
        }
    }
}

// ---------------------------------------------------------------------------
// c1a: (16,1,256,256) f32 -> (16,256,256,64) bf16 NHWC, k5 s1 p2 + ReLU.
// ---------------------------------------------------------------------------
__global__ void conv1a_kernel(const float* __restrict__ in, const float* __restrict__ w,
                              const float* __restrict__ bias, __bf16* __restrict__ out,
                              int total)
{
    int idx = blockIdx.x * blockDim.x + threadIdx.x;
    if (idx >= total) return;
    const int co  = idx & 63;
    const int t   = idx >> 6;
    const int pix = t & 65535;
    const int n   = t >> 16;
    const int oy = pix >> 8, ox = pix & 255;
    float acc = bias[co];
    const float* inN = in + (size_t)n * 65536;
    for (int ky = 0; ky < 5; ++ky) {
        const int iy = oy - 2 + ky;
        if ((unsigned)iy >= 256u) continue;
        for (int kx = 0; kx < 5; ++kx) {
            const int ix = ox - 2 + kx;
            if ((unsigned)ix >= 256u) continue;
            acc += inN[iy * 256 + ix] * w[co * 25 + ky * 5 + kx];
        }
    }
    out[idx] = (__bf16)fmaxf(acc, 0.0f);
}

// ---------------------------------------------------------------------------
// t1a: convT 64->1, k5 s2 p0.  (16,125,125,64) bf16 NHWC -> f32 (16,253,253).
// ---------------------------------------------------------------------------
__global__ void t1a_kernel(const __bf16* __restrict__ in, const float* __restrict__ w,
                           const float* __restrict__ bias, float* __restrict__ out,
                           int total)
{
    int idx = blockIdx.x * blockDim.x + threadIdx.x;
    if (idx >= total) return;
    const int pix = idx % 64009;
    const int n   = idx / 64009;
    const int oy = pix / 253, ox = pix % 253;
    float acc = bias[0];
    const __bf16* inN = in + (size_t)n * 15625 * 64;
    for (int ky = 0; ky < 5; ++ky) {
        const int ty = oy - ky;
        if (ty < 0 || (ty & 1)) continue;
        const int iy = ty >> 1;
        if (iy >= 125) continue;
        for (int kx = 0; kx < 5; ++kx) {
            const int tx = ox - kx;
            if (tx < 0 || (tx & 1)) continue;
            const int ix = tx >> 1;
            if (ix >= 125) continue;
            const __bf16* p = inN + (size_t)(iy * 125 + ix) * 64;
            const int kk = ky * 5 + kx;
#pragma unroll 8
            for (int ci = 0; ci < 64; ++ci)
                acc += (float)p[ci] * w[ci * 25 + kk];
        }
    }
    out[idx] = fmaxf(acc, 0.0f);
}

// ---------------------------------------------------------------------------
// t1b: convT 1->1, k5 s1 p2 on f32 (16,253,253) -> recon.
// ---------------------------------------------------------------------------
__global__ void t1b_kernel(const float* __restrict__ in, const float* __restrict__ w,
                           const float* __restrict__ bias, float* __restrict__ out,
                           int total)
{
    int idx = blockIdx.x * blockDim.x + threadIdx.x;
    if (idx >= total) return;
    const int pix = idx % 64009;
    const int n   = idx / 64009;
    const int oy = pix / 253, ox = pix % 253;
    float acc = bias[0];
    const float* inN = in + (size_t)n * 64009;
    for (int ky = 0; ky < 5; ++ky) {
        const int iy = oy + 2 - ky;
        if ((unsigned)iy >= 253u) continue;
        for (int kx = 0; kx < 5; ++kx) {
            const int ix = ox + 2 - kx;
            if ((unsigned)ix >= 253u) continue;
            acc += inN[iy * 253 + ix] * w[ky * 5 + kx];
        }
    }
    out[idx] = fmaxf(acc, 0.0f);
}

// ---------------------------------------------------------------------------
// VQ, one wave per bottleneck pixel: argmin_k (||e_k||^2 - 2 z.e_k), lanes
// own 4 codes each, shfl_xor min-reduce with smallest-index tie-break.
// ---------------------------------------------------------------------------
__global__ __launch_bounds__(256)
void vq_kernel(const float* __restrict__ z, const float* __restrict__ emb,
               int* __restrict__ idxs, float* __restrict__ after,
               __bf16* __restrict__ after_bf, int total, int HW)
{
    const int gtid = blockIdx.x * blockDim.x + threadIdx.x;
    const int wv   = gtid >> 5;          // one wave per pixel
    const int lane = threadIdx.x & 31;
    if (wv >= total) return;             // wave-uniform exit
    const int n   = wv / HW;
    const int pix = wv - n * HW;

    const float* zp = z + ((size_t)n * EMB_D) * HW + pix;

    float s[4]  = {0.f, 0.f, 0.f, 0.f};
    float ee[4] = {0.f, 0.f, 0.f, 0.f};
    for (int d = 0; d < EMB_D; ++d) {
        const float zd = zp[(size_t)d * HW];          // uniform broadcast load
        const float* ed = emb + d * EMB_K + lane;
#pragma unroll
        for (int j = 0; j < 4; ++j) {
            const float w = ed[32 * j];               // coalesced across lanes
            s[j]  += zd * w;
            ee[j] += w * w;
        }
    }
    float best = ee[0] - 2.0f * s[0];
    int   bk   = lane;
#pragma unroll
    for (int j = 1; j < 4; ++j) {
        const float d2 = ee[j] - 2.0f * s[j];
        const int   k  = lane + 32 * j;
        if (d2 < best) { best = d2; bk = k; }
    }
    for (int off = 16; off > 0; off >>= 1) {
        const float ob = __shfl_xor(best, off, 32);
        const int   ok = __shfl_xor(bk,   off, 32);
        if (ob < best || (ob == best && ok < bk)) { best = ob; bk = ok; }
    }
    if (lane == 0) idxs[wv] = bk;

    float*  ap = after + ((size_t)n * EMB_D) * HW + pix;
    __bf16* ab = after_bf + (size_t)wv * EMB_D;
#pragma unroll
    for (int j = 0; j < 2; ++j) {
        const int d = lane + 32 * j;
        const float v = emb[d * EMB_K + bk];
        ap[(size_t)d * HW] = v;
        ab[d] = (__bf16)v;
    }
}

// ---------------------------------------------------------------------------
extern "C" void kernel_launch(void* const* d_in, const int* in_sizes, int n_in,
                              void* d_out, int out_size, void* d_ws, size_t ws_size,
                              hipStream_t stream)
{
    const float* x     = (const float*)d_in[0];
    const float* emb   = (const float*)d_in[1];
    const float* c1a_w = (const float*)d_in[2];  const float* c1a_b = (const float*)d_in[3];
    const float* c1b_w = (const float*)d_in[4];  const float* c1b_b = (const float*)d_in[5];
    const float* c2a_w = (const float*)d_in[6];  const float* c2a_b = (const float*)d_in[7];
    const float* c2b_w = (const float*)d_in[8];  const float* c2b_b = (const float*)d_in[9];
    const float* c3a_w = (const float*)d_in[10]; const float* c3a_b = (const float*)d_in[11];
    const float* c3b_w = (const float*)d_in[12]; const float* c3b_b = (const float*)d_in[13];
    const float* t3a_w = (const float*)d_in[14]; const float* t3a_b = (const float*)d_in[15];
    const float* t3b_w = (const float*)d_in[16]; const float* t3b_b = (const float*)d_in[17];
    const float* t2a_w = (const float*)d_in[18]; const float* t2a_b = (const float*)d_in[19];
    const float* t2b_w = (const float*)d_in[20]; const float* t2b_b = (const float*)d_in[21];
    const float* t1a_w = (const float*)d_in[22]; const float* t1a_b = (const float*)d_in[23];
    const float* t1b_w = (const float*)d_in[24]; const float* t1b_b = (const float*)d_in[25];

    const int B = 16;

    // d_out slices (return order: recon, idxs, before, after)
    float* recon  = (float*)d_out;                       // 16*253*253 = 1,024,144
    int*   idxs   = (int*)((float*)d_out + 1024144);     // 16*29*29   =    13,456
    float* before = (float*)d_out + 1024144 + 13456;     // 16*64*29*29=   861,184
    float* after  = before + 861184;

    // Workspace layout (bf16 elements)
    const size_t SZ_A = 67108864ull;   // (16,256,256,64)
    const size_t SZ_B = 16257024ull;   // (16,126,126,64)
    const size_t SZ_W = 1945600ull;    // all repacked weights
    if (ws_size < (SZ_A + SZ_B + SZ_W) * sizeof(__bf16)) return;
    __bf16* bufA = (__bf16*)d_ws;
    __bf16* bufB = bufA + SZ_A;
    __bf16* wr   = bufB + SZ_B;

    __bf16* wr_c1b = wr;                  // 64x64   : 102400
    __bf16* wr_c2a = wr_c1b + 102400;     // 64x128  : 204800
    __bf16* wr_c2b = wr_c2a + 204800;     // 128x128 : 409600
    __bf16* wr_c3a = wr_c2b + 409600;     // 128x64  : 204800
    __bf16* wr_c3b = wr_c3a + 204800;     // 64x64   : 102400
    __bf16* wr_t3a = wr_c3b + 102400;     // 64->128 : 204800
    __bf16* wr_t3b = wr_t3a + 204800;     // 128x128 : 409600
    __bf16* wr_t2a = wr_t3b + 409600;     // 128->64 : 204800
    __bf16* wr_t2b = wr_t2a + 204800;     // 64x64   : 102400

    const int NT = 256;
    auto blocks = [](long long n, int t) { return (unsigned)((n + t - 1) / t); };
    auto tiles  = [](int hw) { return (unsigned)((hw + 127) / 128); };  // 4 waves x 32 px

    // ---------------- Weight repack ----------------
    repack_w<<<blocks(102400, NT), NT, 0, stream>>>(c1b_w, wr_c1b, 102400, 64, 64, 0);
    repack_w<<<blocks(204800, NT), NT, 0, stream>>>(c2a_w, wr_c2a, 204800, 64, 128, 0);
    repack_w<<<blocks(409600, NT), NT, 0, stream>>>(c2b_w, wr_c2b, 409600, 128, 128, 0);
    repack_w<<<blocks(204800, NT), NT, 0, stream>>>(c3a_w, wr_c3a, 204800, 128, 64, 0);
    repack_w<<<blocks(102400, NT), NT, 0, stream>>>(c3b_w, wr_c3b, 102400, 64, 64, 0);
    repack_w<<<blocks(204800, NT), NT, 0, stream>>>(t3a_w, wr_t3a, 204800, 64, 128, 1);
    repack_w<<<blocks(409600, NT), NT, 0, stream>>>(t3b_w, wr_t3b, 409600, 128, 128, 1);
    repack_w<<<blocks(204800, NT), NT, 0, stream>>>(t2a_w, wr_t2a, 204800, 128, 64, 1);
    repack_w<<<blocks(102400, NT), NT, 0, stream>>>(t2b_w, wr_t2b, 102400, 64, 64, 1);

    // ---------------- Encoder ----------------
    conv1a_kernel<<<blocks((long long)B * 65536 * 64, NT), NT, 0, stream>>>(
        x, c1a_w, c1a_b, bufA, B * 65536 * 64);
    // c1b: (256,256,64) -> (126,126,64), s2 p0
    conv_wmma<0, 0, 2><<<dim3(tiles(15876), 1, B), 128, 0, stream>>>(
        bufA, wr_c1b, c1b_b, bufB, 64, 64, 256, 256, 126, 126, 0);
    // c2a: -> (126,126,128), s1 p2
    conv_wmma<0, 0, 1><<<dim3(tiles(15876), 2, B), 128, 0, stream>>>(
        bufB, wr_c2a, c2a_b, bufA, 64, 128, 126, 126, 126, 126, 2);
    // c2b: -> (61,61,128), s2 p0
    conv_wmma<0, 0, 2><<<dim3(tiles(3721), 2, B), 128, 0, stream>>>(
        bufA, wr_c2b, c2b_b, bufB, 128, 128, 126, 126, 61, 61, 0);
    // c3a: -> (61,61,64), s1 p2
    conv_wmma<0, 0, 1><<<dim3(tiles(3721), 1, B), 128, 0, stream>>>(
        bufB, wr_c3a, c3a_b, bufA, 128, 64, 61, 61, 61, 61, 2);
    // c3b: -> "before" f32 NCHW (16,64,29,29), s2 p0
    conv_wmma<0, 1, 2><<<dim3(tiles(841), 1, B), 128, 0, stream>>>(
        bufA, wr_c3b, c3b_b, before, 64, 64, 61, 61, 29, 29, 0);

    // ---------------- Vector quantization (1 wave / pixel) ----------------
    vq_kernel<<<blocks((long long)B * 841 * 32, NT), NT, 0, stream>>>(
        before, emb, idxs, after, bufB /* after_bf NHWC */, B * 841, 841);

    // ---------------- Decoder ----------------
    // t3a: (29,29,64) -> (61,61,128), s2 p0
    conv_wmma<1, 0, 2><<<dim3(tiles(3721), 2, B), 128, 0, stream>>>(
        bufB, wr_t3a, t3a_b, bufA, 64, 128, 29, 29, 61, 61, 0);
    // t3b: -> (61,61,128), s1 p2
    conv_wmma<1, 0, 1><<<dim3(tiles(3721), 2, B), 128, 0, stream>>>(
        bufA, wr_t3b, t3b_b, bufB, 128, 128, 61, 61, 61, 61, 2);
    // t2a: -> (125,125,64), s2 p0
    conv_wmma<1, 0, 2><<<dim3(tiles(15625), 1, B), 128, 0, stream>>>(
        bufB, wr_t2a, t2a_b, bufA, 128, 64, 61, 61, 125, 125, 0);
    // t2b: -> (125,125,64), s1 p2
    conv_wmma<1, 0, 1><<<dim3(tiles(15625), 1, B), 128, 0, stream>>>(
        bufA, wr_t2b, t2b_b, bufB, 64, 64, 125, 125, 125, 125, 2);
    // t1a: -> (16,253,253) f32 (Cout=1, scalar); reuse bufA region as f32
    float* d1a = (float*)bufA;
    t1a_kernel<<<blocks((long long)B * 64009, NT), NT, 0, stream>>>(
        bufB, t1a_w, t1a_b, d1a, B * 64009);
    // t1b: -> recon
    t1b_kernel<<<blocks((long long)B * 64009, NT), NT, 0, stream>>>(
        d1a, t1b_w, t1b_b, recon, B * 64009);
}
